// RankAwareSwapPrecision_3135326126283
// MI455X (gfx1250) — matrix operand order repaired and checked by reading, not compiled, not run on hardware
//
#include <hip/hip_runtime.h>
#include <math.h>

// ---------------------------------------------------------------------------
// RankAwareSwap loss, fused for MI455X (gfx1250, wave32, WGP w/ 320KB LDS).
//   K0 (prep): row squared-norms + one-time split of X into bf16 hi/lo planes
//              (Markidis split, ~f32 accuracy on the bf16 WMMA pipe) + zero out.
//   K1: per-16-row-block: WMMA Gram -> sim in LDS (256KB block, never HBM),
//       then small-k iterative selection for the FP/FN rank terms.
// ---------------------------------------------------------------------------

typedef __attribute__((ext_vector_type(16))) __bf16 v16bf;
typedef __attribute__((ext_vector_type(8)))  __bf16 v8bf;
typedef __attribute__((ext_vector_type(8)))  float  v8f;
typedef __attribute__((ext_vector_type(4)))  int    v4i;

#define BATCH   4096
#define DIM     512
#define MARGINF 0.2f
#define KS_MAX  9            // K+1 with K=8
#define POS_INF (3.0e38f)
#define NEG_INF (-3.0e38f)

__device__ __forceinline__ float rankWeight(int r) {
  // w = 1/log2(rank+1) + 1
  return 1.0f / log2f((float)(r + 1)) + 1.0f;
}

// ---- CDNA5 async global->LDS copy (ASYNCcnt path), with safe fallback -----
__device__ __forceinline__ void asyncCopy16(const void* g, void* l) {
#if defined(__gfx1250__) && __has_builtin(__builtin_amdgcn_global_load_async_to_lds_b128)
  // Signature (from clang diagnostic): (global int4*, lds int4*, imm off, cpol)
  void* gn = const_cast<void*>(g);
  __attribute__((address_space(1))) v4i* gp =
      (__attribute__((address_space(1))) v4i*)gn;
  __attribute__((address_space(3))) v4i* lp =
      (__attribute__((address_space(3))) v4i*)l;
  __builtin_amdgcn_global_load_async_to_lds_b128(gp, lp, 0, 0);
#else
  *(uint4*)l = *(const uint4*)g;
#endif
}

__device__ __forceinline__ void asyncWaitAll() {
#if defined(__gfx1250__) && __has_builtin(__builtin_amdgcn_global_load_async_to_lds_b128)
#if __has_builtin(__builtin_amdgcn_s_wait_asynccnt)
  __builtin_amdgcn_s_wait_asynccnt(0);
#else
  asm volatile("s_wait_asynccnt 0x0" ::: "memory");
#endif
#endif
}

// ---------------------------------------------------------------------------
// Kernel 0: sq[i] = ||X_i||^2 ; split X -> Xhi + Xlo (bf16); zero out scalar.
// One wave per row, 8 rows per 256-thread block.
// ---------------------------------------------------------------------------
__global__ void __launch_bounds__(256)
prep_kernel(const float* __restrict__ X, float* __restrict__ sq,
            __bf16* __restrict__ Xhi, __bf16* __restrict__ Xlo,
            float* __restrict__ out) {
  const int row  = blockIdx.x * 8 + (threadIdx.x >> 5);
  const int lane = threadIdx.x & 31;
  const float* xr = X + (size_t)row * DIM;
  float s = 0.f;
  for (int k = lane; k < DIM; k += 32) {
    const float v = xr[k];
    s += v * v;
    const __bf16 h = (__bf16)v;
    Xhi[(size_t)row * DIM + k] = h;
    Xlo[(size_t)row * DIM + k] = (__bf16)(v - (float)h);
  }
#pragma unroll
  for (int off = 16; off; off >>= 1) s += __shfl_xor(s, off, 32);
  if (lane == 0) sq[row] = s;
  if (blockIdx.x == 0 && threadIdx.x == 0) *out = 0.0f;
}

// ---------------------------------------------------------------------------
// Kernel 1: fused Gram (WMMA) + sim + rank-aware selection for 16 rows.
// gridDim.x = 256 (row blocks), blockDim.x = 256 (8 waves).
// ---------------------------------------------------------------------------
__global__ void __launch_bounds__(256, 1)
rank_loss_kernel(const __bf16* __restrict__ Xhi, const __bf16* __restrict__ Xlo,
                 const int* __restrict__ labels, const float* __restrict__ sq,
                 float* __restrict__ out) {
  extern __shared__ char smemRaw[];
  float*  simLds  = (float*)smemRaw;                                   // 16*4096 f32 = 256KB
  int*    lblLds  = (int*)(smemRaw + 16 * BATCH * 4);                  // 4096 i32  = 16KB
  __bf16* hiA     = (__bf16*)(smemRaw + 16 * BATCH * 4 + BATCH * 4);   // 16*512    = 16KB
  __bf16* loA     = (__bf16*)((char*)hiA + 16 * DIM * 2);              // 16*512    = 16KB
  float*  sqiL    = (float*)((char*)loA + 16 * DIM * 2);               // 16 f32
  int*    lblRowL = (int*)((char*)sqiL + 64);                          // 16 i32

  const int tid     = threadIdx.x;
  const int rowBase = blockIdx.x * 16;

  // ---- Stage via async global->LDS: A hi/lo planes (16KB each) + labels ---
  {
    const char* gHi = (const char*)(Xhi + (size_t)rowBase * DIM);
    const char* gLo = (const char*)(Xlo + (size_t)rowBase * DIM);
    const char* gLb = (const char*)labels;
    for (int i = tid; i < 1024; i += 256) {              // 1024 x 16B per plane
      const int b = i * 16;
      asyncCopy16(gHi + b, (char*)hiA + b);
      asyncCopy16(gLo + b, (char*)loA + b);
      asyncCopy16(gLb + b, (char*)lblLds + b);
    }
  }
  if (tid < 16) {
    sqiL[tid]    = sq[rowBase + tid];
    lblRowL[tid] = labels[rowBase + tid];
  }
  asyncWaitAll();
  __syncthreads();

  const int wv   = tid >> 5;     // wave 0..7
  const int lane = tid & 31;
  const int hf   = lane >> 4;    // lane half
  const int l16  = lane & 15;

  // ---- Phase 1: 16x4096 similarity block via split-bf16 WMMA -------------
  for (int t = 0; t < 32; ++t) {
    const int cn = (t << 3) + wv;           // column tile 0..255
    const int c  = cn * 16 + l16;           // this lane's B column / C column
    const __bf16* XhiC = Xhi + (size_t)c * DIM + 16 * hf;
    const __bf16* XloC = Xlo + (size_t)c * DIM + 16 * hf;
    __builtin_prefetch(XhiC, 0, 1);         // global_prefetch_b8 (L2-resident stream)

    v8f acc = {0.f, 0.f, 0.f, 0.f, 0.f, 0.f, 0.f, 0.f};
#pragma unroll 2
    for (int k0 = 0; k0 < DIM; k0 += 32) {
      // A 16x32 bf16 fragment: two contiguous 16B LDS chunks per plane.
      const int aBase = (l16 << 9) + k0 + 16 * hf;
      const v8bf a0h = *(const v8bf*)(hiA + aBase);
      const v8bf a1h = *(const v8bf*)(hiA + aBase + 8);
      const v8bf a0l = *(const v8bf*)(loA + aBase);
      const v8bf a1l = *(const v8bf*)(loA + aBase + 8);
      const v16bf aHi = __builtin_shufflevector(a0h, a1h,
          0,1,2,3,4,5,6,7,8,9,10,11,12,13,14,15);
      const v16bf aLo = __builtin_shufflevector(a0l, a1l,
          0,1,2,3,4,5,6,7,8,9,10,11,12,13,14,15);
      // B 32x16 bf16 fragment: one contiguous 32B global load per plane.
      const v16bf bHi = *(const v16bf*)(XhiC + k0);
      const v16bf bLo = *(const v16bf*)(XloC + k0);
      // Markidis split: G ~= Ahi*Bhi + Ahi*Blo + Alo*Bhi  (~f32 accuracy)
      acc = __builtin_amdgcn_wmma_f32_16x16x32_bf16(false, aHi, false, bHi,
                                                    (short)0, acc, false, false);
      acc = __builtin_amdgcn_wmma_f32_16x16x32_bf16(false, aHi, false, bLo,
                                                    (short)0, acc, false, false);
      acc = __builtin_amdgcn_wmma_f32_16x16x32_bf16(false, aLo, false, bHi,
                                                    (short)0, acc, false, false);
    }

    // Epilogue: sim = -sqrt(max(sq_i - 2G + sq_j, 0)) + (1-match)*MARGIN
    const float sqj  = sq[c];
    const int   lblj = lblLds[c];
#pragma unroll
    for (int vr = 0; vr < 8; ++vr) {
      const int   r    = vr + 8 * hf;      // C layout: VGPR vr -> row vr / vr+8
      const float g    = acc[vr];
      const float res  = sqiL[r] - 2.0f * g + sqj;
      const float dpos = (res > 0.f) ? sqrtf(res) : 0.f;
      const float simv = -dpos + ((lblRowL[r] == lblj) ? 0.f : MARGINF);
      simLds[r * BATCH + c] = simv;
    }
  }
  __syncthreads();

  // ---- Phase 2: per-row small-k selection (replaces double argsort) ------
  float waveAcc = 0.f;
  for (int rr = 0; rr < 2; ++rr) {
    const int    r     = wv * 2 + rr;
    const float* srow  = simLds + (size_t)r * BATCH;
    const int    myLbl = lblRowL[r];

    // positives count -> ks = min(pos, K+1)
    int cnt = 0;
    for (int j = lane; j < BATCH; j += 32) cnt += (lblLds[j] == myLbl) ? 1 : 0;
#pragma unroll
    for (int off = 16; off; off >>= 1) cnt += __shfl_xor(cnt, off, 32);
    const int ks = (cnt < KS_MAX) ? cnt : KS_MAX;

    float lastV = POS_INF; int lastI = -1;
    int   exIdx[KS_MAX];
    int   fpCnt = 0;
    float accFp = 0.f, accFn = 0.f;

    // Top-ks extraction; extraction order == stable rank 1..ks (ties by index).
    for (int tRank = 1; tRank <= ks; ++tRank) {
      float bv = NEG_INF; int bi = -1;
      for (int j = lane; j < BATCH; j += 32) {
        const float s     = srow[j];
        const bool  after = (s < lastV) || (s == lastV && j > lastI);
        if (after && (s > bv || (s == bv && j < bi))) { bv = s; bi = j; }
      }
#pragma unroll
      for (int off = 16; off; off >>= 1) {
        const float ov = __shfl_xor(bv, off, 32);
        const int   oi = __shfl_xor(bi, off, 32);
        if (oi >= 0 && (bi < 0 || ov > bv || (ov == bv && oi < bi))) { bv = ov; bi = oi; }
      }
      exIdx[tRank - 1] = bi;
      lastV = bv; lastI = bi;
      if (bi >= 0 && lblLds[bi] != myLbl) {            // false positive in top-ks
        ++fpCnt;
        accFp += bv * rankWeight(tRank);
      }
    }

    // Top-fpCnt false negatives (same class, rank > ks), with exact ranks.
    lastV = POS_INF; lastI = -1;
    for (int u = 0; u < fpCnt; ++u) {
      float bv = NEG_INF; int bi = -1;
      for (int j = lane; j < BATCH; j += 32) {
        if (lblLds[j] != myLbl) continue;
        bool ex = false;
#pragma unroll
        for (int q = 0; q < KS_MAX; ++q) ex |= (q < ks) && (exIdx[q] == j);
        if (ex) continue;
        const float s     = srow[j];
        const bool  after = (s < lastV) || (s == lastV && j > lastI);
        if (after && (s > bv || (s == bv && j < bi))) { bv = s; bi = j; }
      }
#pragma unroll
      for (int off = 16; off; off >>= 1) {
        const float ov = __shfl_xor(bv, off, 32);
        const int   oi = __shfl_xor(bi, off, 32);
        if (oi >= 0 && (bi < 0 || ov > bv || (ov == bv && oi < bi))) { bv = ov; bi = oi; }
      }
      if (bi < 0) break;
      // exact global rank: 1 + #{elements strictly ahead in stable order}
      int above = 0;
      for (int j = lane; j < BATCH; j += 32) {
        const float s = srow[j];
        above += ((s > bv) || (s == bv && j < bi)) ? 1 : 0;
      }
#pragma unroll
      for (int off = 16; off; off >>= 1) above += __shfl_xor(above, off, 32);
      accFn += bv * rankWeight(above + 1);
      lastV = bv; lastI = bi;
    }

    if (lane == 0) waveAcc += accFp - accFn;
  }
  if (lane == 0) atomicAdd(out, waveAcc);
}

// ---------------------------------------------------------------------------
// Host-side launcher
// ---------------------------------------------------------------------------
extern "C" void kernel_launch(void* const* d_in, const int* in_sizes, int n_in,
                              void* d_out, int out_size, void* d_ws, size_t ws_size,
                              hipStream_t stream) {
  (void)in_sizes; (void)n_in; (void)out_size; (void)ws_size;
  const float* X      = (const float*)d_in[0];   // 4096x512 f32
  const int*   labels = (const int*)d_in[1];     // 4096 i32
  float*       outp   = (float*)d_out;           // scalar f32

  // Workspace layout: sq (16KB) | Xhi (4MB bf16) | Xlo (4MB bf16)
  float*  sqws = (float*)d_ws;
  __bf16* Xhi  = (__bf16*)((char*)d_ws + 16384);
  __bf16* Xlo  = (__bf16*)((char*)d_ws + 16384 + (size_t)BATCH * DIM * 2);

  // K0: row norms + one-time bf16 hi/lo split + zero the output accumulator.
  prep_kernel<<<BATCH / 8, 256, 0, stream>>>(X, sqws, Xhi, Xlo, outp);

  // K1: fused WMMA Gram + selection. Dynamic LDS:
  //   sim 256KB + labels 16KB + hiA 16KB + loA 16KB + stats 128B
  const size_t ldsBytes = (size_t)16 * BATCH * 4 + BATCH * 4
                        + 2 * (16 * DIM * 2) + 128;
  rank_loss_kernel<<<BATCH / 16, 256, ldsBytes, stream>>>(Xhi, Xlo, labels,
                                                          sqws, outp);
}
